// AttentionHead_74929999446645
// MI455X (gfx1250) — matrix operand order repaired and checked by reading, not compiled, not run on hardware
//
#include <hip/hip_runtime.h>
#include <hip/hip_bf16.h>

// ---- vector types for WMMA operands ----
typedef __attribute__((ext_vector_type(16))) __bf16 v16bf;
typedef __attribute__((ext_vector_type(8)))  __bf16 v8bf;
typedef __attribute__((ext_vector_type(8)))  float  v8f;
typedef __attribute__((ext_vector_type(4)))  float  v4f;
typedef __attribute__((ext_vector_type(4)))  int    v4i;

#define WMMA_BF16(A, B, C) \
  __builtin_amdgcn_wmma_f32_16x16x32_bf16(false, (A), false, (B), (short)0, (C), false, false)

#define SHUF16(lo, hi) \
  __builtin_shufflevector((lo), (hi), 0,1,2,3,4,5,6,7,8,9,10,11,12,13,14,15)

// CDNA5 async LDS->global streaming store (ASYNCcnt-tracked), feature-guarded.
#if defined(__has_builtin)
#if __has_builtin(__builtin_amdgcn_global_store_async_from_lds_b128)
#define HAVE_ASYNC_STORE 1
#endif
#endif
#ifndef HAVE_ASYNC_STORE
#define HAVE_ASYNC_STORE 0
#endif

// The builtin takes typed int4 pointers: global (AS1) dst, LDS (AS3) src.
// Generic -> AS-qualified casts go through integers (LDS generic pointers
// carry the LDS offset in the low 32 bits).
typedef __attribute__((address_space(1))) v4i gv4i;
typedef __attribute__((address_space(3))) v4i lv4i;
#define GBL_AS(p) ((gv4i*)(unsigned long long)(p))
#define LDS_AS(p) ((lv4i*)(unsigned int)(unsigned long long)(p))

__device__ inline void wait_async0() {
#if defined(__has_builtin) && __has_builtin(__builtin_amdgcn_s_wait_asynccnt)
  __builtin_amdgcn_s_wait_asynccnt(0);
#else
  asm volatile("s_wait_asynccnt 0x0" ::: "memory");
#endif
}

// Problem constants (B=8, T=2048, E=1024, H=64)
static constexpr int NB = 8;
static constexpr int T  = 2048;
static constexpr int E  = 1024;
static constexpr int H  = 64;
static constexpr int SP = 2052;  // padded LDS row stride (floats); 2052 % 64 == 4 -> bank-skewed

// ----------------------------------------------------------------------------
// Kernel 1: pack Wk/Wq/Wv (f32 [E,H]) into bf16 panels in the WMMA B layout.
// Panel = (w, kt, nt): 32 lanes x 16 bf16.  lane: N = nt*16 + (lane&15),
// K(element e) = kt*32 + (lane>>4)*8 + (e&7) + (e>>3)*16   (two 8-K runs per lane).
// ----------------------------------------------------------------------------
__global__ void pack_w(const float* __restrict__ Wk, const float* __restrict__ Wq,
                       const float* __restrict__ Wv, __bf16* __restrict__ wpk) {
  const int wid  = blockIdx.x * 4 + (threadIdx.x >> 5);  // 0..383
  const int lane = threadIdx.x & 31;
  const int w    = wid >> 7;          // weight index 0..2
  const int rem  = wid & 127;         // kt*4 + nt
  const int kt   = rem >> 2;
  const int nt   = rem & 3;
  const int lh   = lane & 15;
  const int hh   = lane >> 4;
  const float* W = (w == 0) ? Wk : ((w == 1) ? Wq : Wv);

  __bf16 tmp[16];
#pragma unroll
  for (int e = 0; e < 16; ++e) {
    const int K = kt * 32 + hh * 8 + (e & 7) + (e >> 3) * 16;
    const int n = nt * 16 + lh;
    tmp[e] = (__bf16)W[(size_t)K * H + n];
  }
  v8bf lo, hi;
#pragma unroll
  for (int i = 0; i < 8; ++i) { lo[i] = tmp[i]; hi[i] = tmp[8 + i]; }
  __bf16* dst = wpk + (size_t)w * 65536 + (size_t)rem * 512 + (size_t)lane * 16;
  *(v8bf*)dst       = lo;
  *(v8bf*)(dst + 8) = hi;
}

// ----------------------------------------------------------------------------
// Kernel 2: fused q/k/v projection.  One wave computes a 16-row tile of
// (x @ Wk, x @ Wq, x @ Wv).  All 12 B panels of a K-step are preloaded into
// distinct registers so the 24 b128 loads clause up and the s_wait_loadcnt
// staggers across 12 back-to-back WMMAs (instead of wait(0) per tile).
// Outputs: kbf, qbf row-major bf16 [B*T, H];  v transposed bf16 [B, H, T].
// ----------------------------------------------------------------------------
__global__ void proj_qkv(const float* __restrict__ x, const __bf16* __restrict__ wpk,
                         __bf16* __restrict__ kbf, __bf16* __restrict__ qbf,
                         __bf16* __restrict__ vT) {
  const int tile = blockIdx.x * 4 + (threadIdx.x >> 5);  // 0..1023 (B*T/16 tiles)
  const int lane = threadIdx.x & 31;
  const int lh   = lane & 15;
  const int hh   = lane >> 4;
  const int m0   = tile * 16;

  v8f acc[12] = {};
  const float* xr = x + (size_t)(m0 + lh) * E;

  for (int kt = 0; kt < 32; ++kt) {
    // A panel (16-bit A layout): runs [kk, +8) and [kk+16, +8)
    const int kk = kt * 32 + hh * 8;
    const v4f f0 = *(const v4f*)(xr + kk);
    const v4f f1 = *(const v4f*)(xr + kk + 4);
    const v4f f2 = *(const v4f*)(xr + kk + 16);
    const v4f f3 = *(const v4f*)(xr + kk + 20);
    v16bf a;
#pragma unroll
    for (int i = 0; i < 4; ++i) {
      a[i]      = (__bf16)f0[i];
      a[4 + i]  = (__bf16)f1[i];
      a[8 + i]  = (__bf16)f2[i];
      a[12 + i] = (__bf16)f3[i];
    }
    const __bf16* pb = wpk + (size_t)(kt * 4) * 512 + (size_t)lane * 16;
    v16bf bm[12];
#pragma unroll
    for (int w = 0; w < 3; ++w) {
#pragma unroll
      for (int nt = 0; nt < 4; ++nt) {
        const __bf16* bp = pb + (size_t)w * 65536 + (size_t)nt * 512;
        bm[w * 4 + nt] = SHUF16(*(const v8bf*)bp, *(const v8bf*)(bp + 8));
      }
    }
#pragma unroll
    for (int j = 0; j < 12; ++j) acc[j] = WMMA_BF16(a, bm[j], acc[j]);
  }

  // Store from C layout: lane holds N = nt*16+lh, rows m = hh*8 + r.
#pragma unroll
  for (int w = 0; w < 3; ++w) {
#pragma unroll
    for (int nt = 0; nt < 4; ++nt) {
#pragma unroll
      for (int r = 0; r < 8; ++r) {
        const int m = hh * 8 + r;
        const int g = m0 + m;               // global row over B*T
        const int n = nt * 16 + lh;
        const __bf16 val = (__bf16)acc[w * 4 + nt][r];
        if (w == 0)      kbf[(size_t)g * H + n] = val;
        else if (w == 1) qbf[(size_t)g * H + n] = val;
        else {
          const int bb = g >> 11, t = g & (T - 1);
          vT[((size_t)bb * H + n) * T + t] = val;  // [B,H,T]
        }
      }
    }
  }
}

// ----------------------------------------------------------------------------
// Kernel 3: fused attention for one (batch, 16-query tile).
// 256 threads (8 waves). Phase 1: QK^T via WMMA + ALiBi bias + causal mask
// into padded LDS score buffer. Softmax in LDS. The normalized 16x2048 f32
// prob rows (128 KB -- the dominant HBM traffic of the whole problem) are
// streamed to the attn output with GLOBAL_STORE_ASYNC_FROM_LDS_B128: no VGPR
// staging, ASYNCcnt-tracked, and the store stream drains concurrently with
// the phase-2 P@V WMMAs (which only *read* S). Phase 2 reduces partial y
// accumulators across waves with ds_add_f32.
// ----------------------------------------------------------------------------
__global__ void attn_fused(const __bf16* __restrict__ qbf, const __bf16* __restrict__ kbf,
                           const __bf16* __restrict__ vT, const int* __restrict__ is_dec,
                           float* __restrict__ yout, float* __restrict__ attn) {
  extern __shared__ float smem[];
  float* S      = smem;               // 16 * SP
  float* red    = S + 16 * SP;        // 256
  float* rowmax = red + 256;          // 16
  float* rowinv = rowmax + 16;        // 16
  float* yacc   = rowinv + 16;        // 16 * 64

  const int tid  = threadIdx.x;
  const int wave = tid >> 5;
  const int lane = tid & 31;
  const int lh   = lane & 15;
  const int hh   = lane >> 4;
  const int b     = blockIdx.x >> 7;
  const int qbase = (blockIdx.x & 127) * 16;
  const int dec   = *is_dec;

  for (int i = tid; i < 16 * H; i += 256) yacc[i] = 0.f;

  // ---- Phase 1: scores = q @ k^T * H^-0.5 + m*(j-i), causal mask ----
  const __bf16* qr = qbf + (size_t)(b * T + qbase + lh) * H;
  v16bf aq[2];
#pragma unroll
  for (int p = 0; p < 2; ++p) {
    const __bf16* s = qr + p * 32 + hh * 8;
    const v8bf lo = *(const v8bf*)s;
    const v8bf hi = *(const v8bf*)(s + 16);
    aq[p] = SHUF16(lo, hi);
  }

  for (int jt = wave; jt < 128; jt += 8) {
    const __bf16* kr = kbf + (size_t)(b * T + jt * 16 + lh) * H;
    if (jt + 8 < 128)
      __builtin_prefetch(kbf + (size_t)(b * T + (jt + 8) * 16 + lh) * H, 0, 3);
    v8f c = {};
#pragma unroll
    for (int p = 0; p < 2; ++p) {
      const __bf16* s = kr + p * 32 + hh * 8;    // B layout: K runs [hh*8,+8) & [+16,+8)
      const v8bf lo = *(const v8bf*)s;
      const v8bf hi = *(const v8bf*)(s + 16);
      const v16bf bk = SHUF16(lo, hi);
      c = WMMA_BF16(aq[p], bk, c);
    }
    const int kg = jt * 16 + lh;                 // key index (== local column)
#pragma unroll
    for (int r = 0; r < 8; ++r) {
      const int m  = hh * 8 + r;
      const int qg = qbase + m;
      float sc = c[r] * 0.125f + 0.0625f * (float)(kg - qg);
      if (dec && kg > qg) sc = -__builtin_inff();
      S[m * SP + kg] = sc;
    }
  }
  __syncthreads();

  // ---- Softmax over each of the 16 rows (T=2048 wide) ----
  const int row = tid >> 4;
  const int sub = tid & 15;
  float* srow = S + row * SP;

  float pm = -__builtin_inff();
  for (int c = sub; c < T; c += 16) pm = fmaxf(pm, srow[c]);
  red[tid] = pm;
  __syncthreads();
  if (tid < 16) {
    float m = red[tid * 16];
#pragma unroll
    for (int i = 1; i < 16; ++i) m = fmaxf(m, red[tid * 16 + i]);
    rowmax[tid] = m;
  }
  __syncthreads();

  const float rm = rowmax[row];
  float ps = 0.f;
  for (int c = sub; c < T; c += 16) {
    const float e = __expf(srow[c] - rm);
    srow[c] = e;
    ps += e;
  }
  red[tid] = ps;
  __syncthreads();
  if (tid < 16) {
    float s = 0.f;
#pragma unroll
    for (int i = 0; i < 16; ++i) s += red[tid * 16 + i];
    rowinv[tid] = 1.0f / s;
  }
  __syncthreads();

  const float ri = rowinv[row];
  for (int c = sub; c < T; c += 16) srow[c] *= ri;  // normalize in LDS
  __syncthreads();                                  // DS writes visible before async reads

  // ---- Stream attn rows (16 x 8KB f32) from LDS straight to HBM ----
#if HAVE_ASYNC_STORE
#pragma unroll 4
  for (int i = 0; i < 32; ++i) {
    const int flat16 = tid + i * 256;          // 16-byte chunk id, 0..8191
    const int rr     = flat16 >> 9;            // 512 chunks per 8KB row
    const int c16    = flat16 & 511;
    float* lsrc = S + rr * SP + c16 * 4;
    float* gdst = attn + (size_t)(b * T + qbase + rr) * T + c16 * 4;
    __builtin_amdgcn_global_store_async_from_lds_b128(GBL_AS(gdst), LDS_AS(lsrc), 0, 0);
  }
#else
  {
    float* arow = attn + (size_t)(b * T + qbase + row) * T;
    for (int c = sub; c < T; c += 16) arow[c] = srow[c];
  }
#endif

  // ---- Phase 2: y = P @ V (reads S only; async stores drain underneath) ----
  v8f accv[4] = {};
  for (int kt = wave; kt < 64; kt += 8) {
    // A (probs) from LDS, A layout: M = lh, K runs [kt*32+hh*8,+8) & [+16,+8)
    const float* sa = S + lh * SP + kt * 32 + hh * 8;
    const v4f f0 = *(const v4f*)sa;
    const v4f f1 = *(const v4f*)(sa + 4);
    const v4f f2 = *(const v4f*)(sa + 16);
    const v4f f3 = *(const v4f*)(sa + 20);
    v16bf ap;
#pragma unroll
    for (int i = 0; i < 4; ++i) {
      ap[i]      = (__bf16)f0[i];
      ap[4 + i]  = (__bf16)f1[i];
      ap[8 + i]  = (__bf16)f2[i];
      ap[12 + i] = (__bf16)f3[i];
    }
#pragma unroll
    for (int nt = 0; nt < 4; ++nt) {
      // B (v^T) layout: N = h = nt*16+lh, K runs [kt*32+hh*8,+8) & [+16,+8)
      const __bf16* vr = vT + ((size_t)b * H + nt * 16 + lh) * T + kt * 32 + hh * 8;
      const v8bf lo = *(const v8bf*)vr;
      const v8bf hi = *(const v8bf*)(vr + 16);
      const v16bf bv = SHUF16(lo, hi);
      accv[nt] = WMMA_BF16(ap, bv, accv[nt]);
    }
  }
#pragma unroll
  for (int nt = 0; nt < 4; ++nt) {
#pragma unroll
    for (int r = 0; r < 8; ++r) {
      const int m = hh * 8 + r;
      atomicAdd(&yacc[m * H + nt * 16 + lh], accv[nt][r]);   // ds_add_f32
    }
  }
  __syncthreads();

  for (int i = tid; i < 16 * H; i += 256) {
    const int r = i >> 6, h = i & 63;
    yout[(size_t)(b * T + qbase + r) * H + h] = yacc[i];
  }

#if HAVE_ASYNC_STORE
  wait_async0();  // drain async attn stores (S stays live in LDS until here)
#endif
}

// ----------------------------------------------------------------------------
extern "C" void kernel_launch(void* const* d_in, const int* in_sizes, int n_in,
                              void* d_out, int out_size, void* d_ws, size_t ws_size,
                              hipStream_t stream) {
  (void)in_sizes; (void)n_in; (void)out_size; (void)ws_size;
  const float* x  = (const float*)d_in[0];
  const float* Wk = (const float*)d_in[1];
  const float* Wq = (const float*)d_in[2];
  const float* Wv = (const float*)d_in[3];
  const int* isd  = (const int*)d_in[4];

  // Workspace layout (bf16 elements): kbf | qbf | vT | packed weights
  const size_t qkv_elems = (size_t)NB * T * H;          // 1,048,576
  __bf16* kbf = (__bf16*)d_ws;
  __bf16* qbf = kbf + qkv_elems;
  __bf16* vT  = qbf + qkv_elems;
  __bf16* wpk = vT + qkv_elems;                         // 3 * 65536 bf16

  float* yout = (float*)d_out;                          // [B,T,H]
  float* attn = yout + qkv_elems;                       // [B,T,T]

  pack_w  <<<96,  128, 0, stream>>>(Wk, Wq, Wv, wpk);
  proj_qkv<<<256, 128, 0, stream>>>(x, wpk, kbf, qbf, vT);

  const size_t shmem = (size_t)(16 * SP + 256 + 16 + 16 + 16 * H) * sizeof(float);
  attn_fused<<<NB * (T / 16), 256, shmem, stream>>>(qbf, kbf, vT, isd, yout, attn);
}
